// GAT_7713761263970
// MI455X (gfx1250) — matrix operand order repaired and checked by reading, not compiled, not run on hardware
//
#include <hip/hip_runtime.h>
#include <hip/hip_bf16.h>

// ---------------------------------------------------------------------------
// GAT (3-layer, PyG GATConv semantics) for MI455X / gfx1250.
// GEMMs via v_wmma_f32_16x16x32_f16 (f32 accum), LDS staged in fragment order
// so fragment reads are ds_load_b128. Edge softmax/aggregation via L2-resident
// atomics (entire ~160MB working set fits the 192MB L2).
// ---------------------------------------------------------------------------

typedef __attribute__((ext_vector_type(16))) _Float16 v16h;
typedef __attribute__((ext_vector_type(8)))  float    v8f;

#define HEADS 4
#define NEG_SLOPE 0.2f

// ---------------- utility ----------------
__global__ void zero_f32(float* __restrict__ p, long n) {
    long i = (long)blockIdx.x * blockDim.x + threadIdx.x;
    if (i < n) p[i] = 0.0f;
}

// monotone float -> uint key (order-preserving); key(any finite) > 0
__device__ __forceinline__ unsigned fkey(float f) {
    unsigned u = __float_as_uint(f);
    return (u & 0x80000000u) ? ~u : (u | 0x80000000u);
}
__device__ __forceinline__ float funkey(unsigned k) {
    unsigned u = (k & 0x80000000u) ? (k ^ 0x80000000u) : ~k;
    return __uint_as_float(u);
}
__device__ __forceinline__ float lrelu(float x) {
    return x > 0.0f ? x : NEG_SLOPE * x;
}

// ---------------- WMMA GEMM: Out[M,Ncols] (+)= A[M,K] x B[K,Ncols] ----------
// Block: 256 threads = 8 waves; block tile 32(M) x 128(N); K-step 32.
// LDS holds tiles pre-permuted into WMMA fragment order: each lane's 16
// halves are contiguous (ds_load_b128 x2 per fragment). Each wave computes
// two 16x16 tiles per K-step (A reuse, 2x v_wmma per iteration).
// ACCUM is a compile-time flag; full-tile blocks take a branch-free epilogue.
template <bool ACCUM>
__global__ void __launch_bounds__(256)
gemm_wmma_f16(const float* __restrict__ Ag, const float* __restrict__ Bg,
              float* __restrict__ Out, int M, int K, int Ncols) {
    __shared__ v16h Afr[2 * 32];   // [mtile][lane] -> 16 contiguous halves
    __shared__ v16h Bfr[8 * 32];   // [ntile][lane]
    unsigned* AlU = (unsigned*)Afr;
    unsigned* BlU = (unsigned*)Bfr;

    const int t    = threadIdx.x;
    const int lane = t & 31;
    const int w    = t >> 5;        // wave 0..7
    const int wm   = w >> 2;        // m-tile 0..1
    const int wn0  = (w & 3) * 2;   // first of two n-tiles
    const int g    = lane >> 4;
    const int l16  = lane & 15;
    const int bM   = blockIdx.y * 32;
    const int bN   = blockIdx.x * 128;

    union Acc { v8f v; float e[8]; };
    Acc acc0, acc1;
    acc0.v = {}; acc1.v = {};

    for (int k0 = 0; k0 < K; k0 += 32) {
        // ---- stage A 32x32: pack (k,k+1) f32->2xf16, store u32 at fragment
        //      slot. A-frag map: i = ((k>>4)<<3)+(k&7), lane = (m&15)+16*((k>>3)&1)
        #pragma unroll
        for (int pj = 0; pj < 2; ++pj) {
            int p  = t + 256 * pj;         // 0..511 payloads
            int m  = p >> 4;               // 0..31
            int k  = (p & 15) << 1;        // even 0..30
            int gr = bM + m, gk = k0 + k;
            float f0 = Ag[(long)min(gr, M - 1) * K + min(gk,     K - 1)];
            float f1 = Ag[(long)min(gr, M - 1) * K + min(gk + 1, K - 1)];
            f0 = (gr < M && gk     < K) ? f0 : 0.0f;
            f1 = (gr < M && gk + 1 < K) ? f1 : 0.0f;
            union { _Float16 h[2]; unsigned u; } pk;
            pk.h[0] = (_Float16)f0; pk.h[1] = (_Float16)f1;
            int gA = (k >> 3) & 1;
            int iA = ((k >> 4) << 3) + (k & 7);        // even
            int mt = m >> 4, mloc = m & 15;
            AlU[(mt * 32 + mloc + 16 * gA) * 8 + (iA >> 1)] = pk.u;
        }
        // ---- stage B 32x128: B-frag map: i = ((k>>4)<<3)+(k&7), lane = (n&15)+16*((k>>3)&1)
        #pragma unroll
        for (int pj = 0; pj < 8; ++pj) {
            int p  = t + 256 * pj;         // 0..2047 payloads
            int n  = p & 127;
            int k  = (p >> 7) << 1;        // even 0..30
            int gk = k0 + k, gc = bN + n;
            float f0 = Bg[(long)min(gk,     K - 1) * Ncols + min(gc, Ncols - 1)];
            float f1 = Bg[(long)min(gk + 1, K - 1) * Ncols + min(gc, Ncols - 1)];
            f0 = (gk     < K && gc < Ncols) ? f0 : 0.0f;
            f1 = (gk + 1 < K && gc < Ncols) ? f1 : 0.0f;
            union { _Float16 h[2]; unsigned u; } pk;
            pk.h[0] = (_Float16)f0; pk.h[1] = (_Float16)f1;
            int gB = (k >> 3) & 1;
            int iB = ((k >> 4) << 3) + (k & 7);        // even
            int nt = n >> 4, nloc = n & 15;
            BlU[(nt * 32 + nloc + 16 * gB) * 8 + (iB >> 1)] = pk.u;
        }
        __syncthreads();

        v16h a  = Afr[wm * 32 + lane];          // 2x ds_load_b128
        v16h b0 = Bfr[wn0 * 32 + lane];
        v16h b1 = Bfr[(wn0 + 1) * 32 + lane];
        acc0.v = __builtin_amdgcn_wmma_f32_16x16x32_f16(
            false, a, false, b0, (short)0, acc0.v, false, false);
        acc1.v = __builtin_amdgcn_wmma_f32_16x16x32_f16(
            false, a, false, b1, (short)0, acc1.v, false, false);
        __syncthreads();
    }

    // C/D layout: reg r, lanes0-15 -> M=r, lanes16-31 -> M=r+8
    const int col0 = bN + wn0 * 16 + l16;
    const int col1 = col0 + 16;
    if (bM + 32 <= M && bN + 128 <= Ncols) {
        // full tile: branch-free epilogue (uniform condition)
        #pragma unroll
        for (int r = 0; r < 8; ++r) {
            long o0 = (long)(bM + wm * 16 + r + g * 8) * Ncols + col0;
            if (ACCUM) { Out[o0] += acc0.e[r]; Out[o0 + 16] += acc1.e[r]; }
            else       { Out[o0]  = acc0.e[r]; Out[o0 + 16]  = acc1.e[r]; }
        }
    } else {
        #pragma unroll
        for (int r = 0; r < 8; ++r) {
            int row = bM + wm * 16 + r + g * 8;
            if (row < M) {
                if (col0 < Ncols) {
                    long o = (long)row * Ncols + col0;
                    if (ACCUM) Out[o] += acc0.e[r]; else Out[o] = acc0.e[r];
                }
                if (col1 < Ncols) {
                    long o = (long)row * Ncols + col1;
                    if (ACCUM) Out[o] += acc1.e[r]; else Out[o] = acc1.e[r];
                }
            }
        }
    }
}

// ---------------- per-node attention dots: a_s/a_d [N,H] --------------------
__global__ void attn_dots(const float* __restrict__ h,
                          const float* __restrict__ att_s,
                          const float* __restrict__ att_d,
                          float* __restrict__ aS, float* __restrict__ aD,
                          int Nn, int C) {
    int tid = blockIdx.x * blockDim.x + threadIdx.x;
    if (tid >= Nn * HEADS) return;
    int n = tid / HEADS, hh = tid % HEADS;
    const float* hp = h + (long)n * (HEADS * C) + hh * C;
    const float* as = att_s + hh * C;
    const float* ad = att_d + hh * C;
    float s = 0.0f, d = 0.0f;
    for (int c = 0; c < C; ++c) { float v = hp[c]; s += v * as[c]; d += v * ad[c]; }
    aS[tid] = s; aD[tid] = d;
}

// ---------------- edge pass 1: segment max via ordered-uint atomicMax -------
__global__ void edge_max(const int* __restrict__ ei, int E, int Nn,
                         const float* __restrict__ aS, const float* __restrict__ aD,
                         unsigned* __restrict__ mkey) {
    int tid = blockIdx.x * blockDim.x + threadIdx.x;
    int Et = E + Nn;
    if (tid >= Et * HEADS) return;
    int e = tid / HEADS, hh = tid % HEADS;
    int src = (e < E) ? ei[e] : (e - E);
    int dst = (e < E) ? ei[E + e] : (e - E);
    float sc = lrelu(aS[src * HEADS + hh] + aD[dst * HEADS + hh]);
    atomicMax(&mkey[dst * HEADS + hh], fkey(sc));
}

// ---------------- edge pass 2: denom = segment_sum(exp(e - m)) --------------
__global__ void edge_sum(const int* __restrict__ ei, int E, int Nn,
                         const float* __restrict__ aS, const float* __restrict__ aD,
                         const unsigned* __restrict__ mkey,
                         float* __restrict__ denom) {
    int tid = blockIdx.x * blockDim.x + threadIdx.x;
    int Et = E + Nn;
    if (tid >= Et * HEADS) return;
    int e = tid / HEADS, hh = tid % HEADS;
    int src = (e < E) ? ei[e] : (e - E);
    int dst = (e < E) ? ei[E + e] : (e - E);
    float sc = lrelu(aS[src * HEADS + hh] + aD[dst * HEADS + hh]);
    float m  = funkey(mkey[dst * HEADS + hh]);
    atomicAdd(&denom[dst * HEADS + hh], __expf(sc - m));
}

// ---------------- edge pass 3 (HC=256): agg[dst] += alpha * h[src] ----------
// One wave per edge; lane owns 8 consecutive channels (head uniform per lane)
// -> gather is 2x global_load_b128 per lane, 1KB contiguous per wave.
__global__ void __launch_bounds__(256)
edge_aggregate_256(const int* __restrict__ ei, int E, int Nn,
                   const float* __restrict__ aS, const float* __restrict__ aD,
                   const unsigned* __restrict__ mkey, const float* __restrict__ denom,
                   const float* __restrict__ h, float* __restrict__ agg) {
    const int lane = threadIdx.x & 31;
    const int e    = blockIdx.x * 8 + (threadIdx.x >> 5);
    if (e >= E + Nn) return;
    int src = (e < E) ? ei[e] : (e - E);
    int dst = (e < E) ? ei[E + e] : (e - E);
    const int cbase = lane * 8;          // channels [cbase, cbase+8)
    const int hh    = lane >> 3;         // head = cbase/64, uniform per lane
    float sc = lrelu(aS[src * HEADS + hh] + aD[dst * HEADS + hh]);
    float m  = funkey(mkey[dst * HEADS + hh]);
    float alpha = __expf(sc - m) / denom[dst * HEADS + hh];
    const float4* hs = (const float4*)(h + (long)src * 256 + cbase);
    float*        ag = agg + (long)dst * 256 + cbase;
    float4 v0 = hs[0], v1 = hs[1];
    atomicAdd(&ag[0], alpha * v0.x); atomicAdd(&ag[1], alpha * v0.y);
    atomicAdd(&ag[2], alpha * v0.z); atomicAdd(&ag[3], alpha * v0.w);
    atomicAdd(&ag[4], alpha * v1.x); atomicAdd(&ag[5], alpha * v1.y);
    atomicAdd(&ag[6], alpha * v1.z); atomicAdd(&ag[7], alpha * v1.w);
}

// ---------------- edge pass 3 (generic C): coalesced c = lane + 32j ---------
__global__ void __launch_bounds__(256)
edge_aggregate(const int* __restrict__ ei, int E, int Nn,
               const float* __restrict__ aS, const float* __restrict__ aD,
               const unsigned* __restrict__ mkey, const float* __restrict__ denom,
               const float* __restrict__ h, float* __restrict__ agg, int C) {
    const int lane = threadIdx.x & 31;
    const int e    = blockIdx.x * 8 + (threadIdx.x >> 5);
    if (e >= E + Nn) return;
    int src = (e < E) ? ei[e] : (e - E);
    int dst = (e < E) ? ei[E + e] : (e - E);
    const int HC = HEADS * C;
    const float* hs = h   + (long)src * HC;
    float*       ag = agg + (long)dst * HC;
    for (int c = lane; c < HC; c += 32) {
        int hh = c / C;
        float sc = lrelu(aS[src * HEADS + hh] + aD[dst * HEADS + hh]);
        float m  = funkey(mkey[dst * HEADS + hh]);
        float alpha = __expf(sc - m) / denom[dst * HEADS + hh];
        atomicAdd(&ag[c], alpha * hs[c]);
    }
}

// ---------------- combine for layers 1,2: elu(agg + b + sb) -----------------
// (agg already contains the skip GEMM, accumulated in-place)
__global__ void combine_elu(const float* __restrict__ agg,
                            const float* __restrict__ b, const float* __restrict__ sb,
                            float* __restrict__ out, long total, int HC) {
    long i = (long)blockIdx.x * blockDim.x + threadIdx.x;
    if (i >= total) return;
    int k = (int)(i % HC);
    float v = agg[i] + b[k] + sb[k];
    out[i] = v > 0.0f ? v : (__expf(v) - 1.0f);
}

// ---------------- final: head-mean + skip + bias -> log_softmax(47) ---------
__global__ void __launch_bounds__(256)
final_logsoftmax(const float* __restrict__ agg,   // [N, 4*47]
                 const float* __restrict__ skip,  // [N, 47]
                 const float* __restrict__ b, const float* __restrict__ sb,
                 float* __restrict__ out, int Nn) {
    const int lane = threadIdx.x & 31;
    const int n    = blockIdx.x * 8 + (threadIdx.x >> 5);
    if (n >= Nn) return;
    const int OUT = 47;
    const float* ag = agg + (long)n * (HEADS * OUT);
    const float* sk = skip + (long)n * OUT;

    int c1 = lane, c2 = lane + 32;
    bool has2 = c2 < OUT;
    float v1, v2 = 0.0f;
    {
        float s = 0.0f;
        #pragma unroll
        for (int hh = 0; hh < HEADS; ++hh) s += ag[hh * OUT + c1];
        v1 = 0.25f * s + b[c1] + sk[c1] + sb[c1];
    }
    if (has2) {
        float s = 0.0f;
        #pragma unroll
        for (int hh = 0; hh < HEADS; ++hh) s += ag[hh * OUT + c2];
        v2 = 0.25f * s + b[c2] + sk[c2] + sb[c2];
    }
    float mx = has2 ? fmaxf(v1, v2) : v1;
    #pragma unroll
    for (int off = 16; off > 0; off >>= 1)
        mx = fmaxf(mx, __shfl_xor(mx, off, 32));
    float se = __expf(v1 - mx) + (has2 ? __expf(v2 - mx) : 0.0f);
    #pragma unroll
    for (int off = 16; off > 0; off >>= 1)
        se += __shfl_xor(se, off, 32);
    float L = mx + logf(se);
    out[(long)n * OUT + c1] = v1 - L;
    if (has2) out[(long)n * OUT + c2] = v2 - L;
}

// ---------------------------------------------------------------------------
extern "C" void kernel_launch(void* const* d_in, const int* in_sizes, int n_in,
                              void* d_out, int out_size, void* d_ws, size_t ws_size,
                              hipStream_t stream) {
    const int F_IN = 100, HID = 64, OUT = 47, HC = HEADS * HID;   // 256
    const int Nn = in_sizes[0] / F_IN;                             // 50000
    const int E  = in_sizes[1] / 2;                                // 500000

    const float* x   = (const float*)d_in[0];
    const int*   ei  = (const int*)  d_in[1];
    const float* w1  = (const float*)d_in[2];
    const float* as1 = (const float*)d_in[3];
    const float* ad1 = (const float*)d_in[4];
    const float* b1  = (const float*)d_in[5];
    const float* sw1 = (const float*)d_in[6];
    const float* sb1 = (const float*)d_in[7];
    const float* w2  = (const float*)d_in[8];
    const float* as2 = (const float*)d_in[9];
    const float* ad2 = (const float*)d_in[10];
    const float* b2  = (const float*)d_in[11];
    const float* sw2 = (const float*)d_in[12];
    const float* sb2 = (const float*)d_in[13];
    const float* w3  = (const float*)d_in[14];
    const float* as3 = (const float*)d_in[15];
    const float* ad3 = (const float*)d_in[16];
    const float* b3  = (const float*)d_in[17];
    const float* sw3 = (const float*)d_in[18];
    const float* sb3 = (const float*)d_in[19];
    float* outp = (float*)d_out;

    // workspace layout (floats); total ~166 MB — node-resident, L2-sized
    float*    A     = (float*)d_ws;                 // layer input  [N,256]
    float*    B     = A + (size_t)Nn * HC;          // h = in @ W   [N,256]
    float*    D     = B + (size_t)Nn * HC;          // agg (+skip)  [N,256]
    float*    aS    = D + (size_t)Nn * HC;          // [N,4]
    float*    aD    = aS + (size_t)Nn * HEADS;      // [N,4]
    unsigned* mkey  = (unsigned*)(aD + (size_t)Nn * HEADS);  // [N,4]
    float*    denom = (float*)mkey + (size_t)Nn * HEADS;     // [N,4]
    float*    skip3 = denom + (size_t)Nn * HEADS;   // [N,47]

    const int Et = E + Nn;                          // edges incl. self-loops
    const int TB = 256;
    const dim3 blk(TB);
    auto gemm_grid = [&](int cols) { return dim3((cols + 127) / 128, (Nn + 31) / 32); };
    const int edgeThreads = (Et * HEADS + TB - 1) / TB;
    const int edgeWaves   = (Et + 7) / 8;
    const int nodeWaves   = (Nn + 7) / 8;

    // ---------------- layer 1 (F_IN=100 -> HC=256) ----------------
    zero_f32<<<((long)Nn * HC + TB - 1) / TB, blk, 0, stream>>>(D, (long)Nn * HC);
    zero_f32<<<((long)Nn * HEADS * 2 + TB - 1) / TB, blk, 0, stream>>>((float*)mkey, (long)Nn * HEADS * 2);
    gemm_wmma_f16<false><<<gemm_grid(HC), blk, 0, stream>>>(x, w1,  B, Nn, F_IN, HC);
    gemm_wmma_f16<true ><<<gemm_grid(HC), blk, 0, stream>>>(x, sw1, D, Nn, F_IN, HC);
    attn_dots<<<(Nn * HEADS + TB - 1) / TB, blk, 0, stream>>>(B, as1, ad1, aS, aD, Nn, HID);
    edge_max<<<edgeThreads, blk, 0, stream>>>(ei, E, Nn, aS, aD, mkey);
    edge_sum<<<edgeThreads, blk, 0, stream>>>(ei, E, Nn, aS, aD, mkey, denom);
    edge_aggregate_256<<<edgeWaves, blk, 0, stream>>>(ei, E, Nn, aS, aD, mkey, denom, B, D);
    combine_elu<<<((long)Nn * HC + TB - 1) / TB, blk, 0, stream>>>(D, b1, sb1, A, (long)Nn * HC, HC);

    // ---------------- layer 2 (256 -> 256) ----------------
    zero_f32<<<((long)Nn * HC + TB - 1) / TB, blk, 0, stream>>>(D, (long)Nn * HC);
    zero_f32<<<((long)Nn * HEADS * 2 + TB - 1) / TB, blk, 0, stream>>>((float*)mkey, (long)Nn * HEADS * 2);
    gemm_wmma_f16<false><<<gemm_grid(HC), blk, 0, stream>>>(A, w2,  B, Nn, HC, HC);
    gemm_wmma_f16<true ><<<gemm_grid(HC), blk, 0, stream>>>(A, sw2, D, Nn, HC, HC);
    attn_dots<<<(Nn * HEADS + TB - 1) / TB, blk, 0, stream>>>(B, as2, ad2, aS, aD, Nn, HID);
    edge_max<<<edgeThreads, blk, 0, stream>>>(ei, E, Nn, aS, aD, mkey);
    edge_sum<<<edgeThreads, blk, 0, stream>>>(ei, E, Nn, aS, aD, mkey, denom);
    edge_aggregate_256<<<edgeWaves, blk, 0, stream>>>(ei, E, Nn, aS, aD, mkey, denom, B, D);
    combine_elu<<<((long)Nn * HC + TB - 1) / TB, blk, 0, stream>>>(D, b2, sb2, A, (long)Nn * HC, HC);

    // ---------------- layer 3 (256 -> 4x47, mean heads, +skip, log_softmax) -
    const int HO = HEADS * OUT;  // 188
    zero_f32<<<((long)Nn * HO + TB - 1) / TB, blk, 0, stream>>>(D, (long)Nn * HO);
    zero_f32<<<((long)Nn * HEADS * 2 + TB - 1) / TB, blk, 0, stream>>>((float*)mkey, (long)Nn * HEADS * 2);
    gemm_wmma_f16<false><<<gemm_grid(HO),  blk, 0, stream>>>(A, w3,  B,     Nn, HC, HO);
    gemm_wmma_f16<false><<<gemm_grid(OUT), blk, 0, stream>>>(A, sw3, skip3, Nn, HC, OUT);
    attn_dots<<<(Nn * HEADS + TB - 1) / TB, blk, 0, stream>>>(B, as3, ad3, aS, aD, Nn, OUT);
    edge_max<<<edgeThreads, blk, 0, stream>>>(ei, E, Nn, aS, aD, mkey);
    edge_sum<<<edgeThreads, blk, 0, stream>>>(ei, E, Nn, aS, aD, mkey, denom);
    edge_aggregate<<<edgeWaves, blk, 0, stream>>>(ei, E, Nn, aS, aD, mkey, denom, B, D, OUT);
    final_logsoftmax<<<nodeWaves, blk, 0, stream>>>(D, skip3, b3, sb3, outp, Nn);
}
